// PaliGemmaWithExpertModel_58884001628585
// MI455X (gfx1250) — compile-verified, matrix operand order checked by reading, not run on hardware
//
#include <hip/hip_runtime.h>
#include <hip/hip_bf16.h>

typedef __attribute__((ext_vector_type(16))) __bf16 v16bf;
typedef __attribute__((ext_vector_type(8)))  __bf16 v8bf;
typedef __attribute__((ext_vector_type(8)))  float  v8f;

typedef __attribute__((ext_vector_type(4))) unsigned int tdm_g0_t;
typedef __attribute__((ext_vector_type(8))) int          tdm_g1_t;
typedef __attribute__((ext_vector_type(4))) int          tdm_g2_t;

#if defined(__has_builtin)
#  if __has_builtin(__builtin_amdgcn_tensor_load_to_lds) && \
      __has_builtin(__builtin_amdgcn_s_wait_tensorcnt)
#    define ATHENA_HAVE_TDM 1
#  endif
#endif
#ifndef ATHENA_HAVE_TDM
#  define ATHENA_HAVE_TDM 0
#endif

namespace {
constexpr int   kB   = 4;
constexpr int   kL1  = 768;
constexpr int   kL2  = 50;
constexpr int   kL   = 818;
constexpr int   kD   = 2048;
constexpr int   kH   = 8;
constexpr int   kHD  = 256;
constexpr int   kMV  = 8192;
constexpr int   kME  = 4096;
constexpr float kEps   = 1e-6f;
constexpr float kScale = 0.0625f;   // 256^-0.5
constexpr float kNegBig = -1e30f;
}

__device__ __forceinline__ v8f wmma_bf16(v16bf a, v16bf b, v8f c) {
  return __builtin_amdgcn_wmma_f32_16x16x32_bf16(false, a, false, b, (short)0, c,
                                                 false, false);
}

// 16-bit WMMA fragment from a contiguous 32-element line (CDNA5 ISA):
// lanes 0-15 take [half*8 .. +7] and [16+half*8 .. +7]; half = lane>>4.
__device__ __forceinline__ v16bf frag_from(const __bf16* line32, int half) {
  v8bf lo = *(const v8bf*)(line32 + half * 8);
  v8bf hi = *(const v8bf*)(line32 + 16 + half * 8);
  return __builtin_shufflevector(lo, hi, 0, 1, 2, 3, 4, 5, 6, 7, 8, 9, 10, 11, 12,
                                 13, 14, 15);
}

__device__ __forceinline__ v8bf cvt8(const float* p) {
  float4 a = *(const float4*)p;
  float4 b = *(const float4*)(p + 4);
  v8bf r;
  r[0] = (__bf16)a.x; r[1] = (__bf16)a.y; r[2] = (__bf16)a.z; r[3] = (__bf16)a.w;
  r[4] = (__bf16)b.x; r[5] = (__bf16)b.y; r[6] = (__bf16)b.z; r[7] = (__bf16)b.w;
  return r;
}

__device__ __forceinline__ float gelu_tanh(float x) {
  return 0.5f * x * (1.0f + tanhf(0.7978845608f * (x + 0.044715f * x * x * x)));
}

// Issue a 2D TDM tile load (D# per CDNA5 ISA, 2-byte elems).
// OOB reads (X >= tensor_d0 or Y >= tensor_d1) are zero-filled by hardware.
__device__ __forceinline__ void tdm_load_2d(const __bf16* gsrc, unsigned lds_addr,
                                            unsigned tensor_d0, unsigned tensor_d1,
                                            unsigned tile_d0, unsigned tile_d1,
                                            unsigned stride_d0) {
#if ATHENA_HAVE_TDM
  unsigned long long ga = (unsigned long long)(size_t)gsrc;
  tdm_g0_t g0;
  g0[0] = 1u;                                              // count=1
  g0[1] = lds_addr;                                        // lds_addr
  g0[2] = (unsigned)(ga & 0xffffffffu);                    // global_addr lo
  g0[3] = (unsigned)((ga >> 32) & 0x1ffffffu) | (2u << 30);// hi | type=2
  tdm_g1_t g1;
  g1[0] = (int)(1u << 16);                                 // data_size = 2B
  g1[1] = (int)((tensor_d0 & 0xffffu) << 16);
  g1[2] = (int)(((tensor_d0 >> 16) & 0xffffu) | ((tensor_d1 & 0xffffu) << 16));
  g1[3] = (int)(((tensor_d1 >> 16) & 0xffffu) | ((tile_d0 & 0xffffu) << 16));
  g1[4] = (int)(tile_d1 & 0xffffu);                        // tile_dim1, tile_dim2=0
  g1[5] = (int)stride_d0;                                  // dim0 stride lo32
  g1[6] = 0;
  g1[7] = 0;
  tdm_g2_t gz;
  gz[0] = 0; gz[1] = 0; gz[2] = 0; gz[3] = 0;
#if __clang_major__ >= 23
  tdm_g1_t gz8;
#pragma unroll
  for (int i = 0; i < 8; ++i) gz8[i] = 0;
  __builtin_amdgcn_tensor_load_to_lds(g0, g1, gz, gz, gz8, 0);
#else
  __builtin_amdgcn_tensor_load_to_lds(g0, g1, gz, gz, 0);
#endif
#else
  (void)gsrc; (void)lds_addr; (void)tensor_d0; (void)tensor_d1;
  (void)tile_d0; (void)tile_d1; (void)stride_d0;
#endif
}

// ---------------------------------------------------------------------------
__global__ void concat_init_kernel(const float* __restrict__ pre,
                                   const float* __restrict__ suf,
                                   float* __restrict__ h) {
  long idx = (long)blockIdx.x * blockDim.x + threadIdx.x;
  long total = (long)kB * kL * kD;
  if (idx >= total) return;
  int d = (int)(idx % kD);
  long r = idx / kD;
  int t = (int)(r % kL);
  int b = (int)(r / kL);
  float v;
  if (t < kL1) v = pre[((long)b * kL1 + t) * kD + d];
  else         v = suf[((long)b * kL2 + (t - kL1)) * kD + d];
  h[idx] = v;
}

// ---------------------------------------------------------------------------
__global__ __launch_bounds__(256) void rmsnorm_kernel(const float* __restrict__ h,
                                                      const float* __restrict__ wp,
                                                      const float* __restrict__ we,
                                                      __bf16* __restrict__ out) {
  __shared__ float red[256];
  int row = blockIdx.x;                 // 0 .. B*L-1
  int t = row % kL;
  const float* w = (t < kL1) ? wp : we;
  const float* x = h + (long)row * kD;
  float s = 0.f;
  for (int i = threadIdx.x; i < kD; i += 256) {
    float v = x[i];
    s += v * v;
  }
  red[threadIdx.x] = s;
  __syncthreads();
  for (int off = 128; off > 0; off >>= 1) {
    if (threadIdx.x < off) red[threadIdx.x] += red[threadIdx.x + off];
    __syncthreads();
  }
  float rs = rsqrtf(red[0] * (1.0f / kD) + kEps);
  for (int i = threadIdx.x; i < kD; i += 256)
    out[(long)row * kD + i] = (__bf16)(x[i] * rs * (1.f + w[i]));
}

// ---------------------------------------------------------------------------
// bf16 WMMA GEMM, block tile 64(M) x 128(N), BK = 32, 8 waves.
// A tile via TDM when use_tdm (prefix launches; tiles never straddle a batch).
// W tiles staged transposed so every B fragment is 2 x ds_load_b128.
// mode 0: outF = C      mode 1: outF += C
// mode 3: outB[b][n][t] = bf16(C)   (transposed store, feeds attention V tile)
// DUAL: two GEMMs sharing A; outB = bf16(gelu(C1)*C2).
// ---------------------------------------------------------------------------
template <bool DUAL>
__global__ __launch_bounds__(256) void gemm_bf16_kernel(
    const __bf16* __restrict__ A, int lda,
    const float* __restrict__ W1, const float* __restrict__ W2, int ldw,
    float* __restrict__ outF, __bf16* __restrict__ outB, int ldo,
    int Mrows, int K, int Ltok, int tok_off, int use_tdm, int mode) {
  __shared__ alignas(16) __bf16 sA[64 * 32];      // row-major tile
  __shared__ alignas(16) __bf16 sBt1[128 * 32];   // transposed: [n][k]
  __shared__ alignas(16) __bf16 sBt2[128 * 32];

  const int tid  = threadIdx.x;
  const int lane = tid & 31;
  const int wid  = tid >> 5;
  const int wm   = wid >> 1;
  const int wn   = wid & 1;
  const int half = lane >> 4;
  const int lr   = lane & 15;
  const int m0   = blockIdx.x * 64;
  const int n0   = blockIdx.y * 128;

  v8f zf = {};
  v8f acc[4], acc2[4];
#pragma unroll
  for (int i = 0; i < 4; ++i) { acc[i] = zf; acc2[i] = zf; }

  const int a_row = (tid * 8) >> 5;     // 0..63
  const int a_col = (tid * 8) & 31;     // multiple of 8
  long a_g;
  {
    int gm = m0 + a_row;
    if (gm >= Mrows) gm = Mrows - 1;    // clamped rows never stored
    a_g = (long)(gm / Ltok) * kL + tok_off + (gm % Ltok);
  }
  const int bt_n  = tid & 127;
  const int bt_kh = (tid >> 7) * 16;    // 0 or 16

  bool tdm_active = false;
#if ATHENA_HAVE_TDM
  tdm_active = (use_tdm != 0);
  long tdm_g0row = 0;
  if (tdm_active) {
    tdm_g0row = (long)(m0 / Ltok) * kL + tok_off + (m0 % Ltok);
  }
#endif

  for (int k0 = 0; k0 < K; k0 += 32) {
    // ---- stage A tile ----
    if (!tdm_active) {
      *(v8bf*)(sA + a_row * 32 + a_col) =
          *(const v8bf*)(A + a_g * lda + k0 + a_col);
    }
#if ATHENA_HAVE_TDM
    else if (wid == 0) {
      tdm_load_2d(A + tdm_g0row * lda + k0, (unsigned)(size_t)(&sA[0]),
                  (unsigned)lda, 64u, 32u, 64u, (unsigned)lda);
      __builtin_amdgcn_s_wait_tensorcnt(0);
    }
#endif
    // ---- stage W tile(s), transposed, coalesced b32 loads ----
    {
      const float* src = W1 + (long)(k0 + bt_kh) * ldw + n0 + bt_n;
      v8bf pk0, pk1;
#pragma unroll
      for (int j = 0; j < 8; ++j) pk0[j] = (__bf16)src[(long)j * ldw];
#pragma unroll
      for (int j = 0; j < 8; ++j) pk1[j] = (__bf16)src[(long)(8 + j) * ldw];
      *(v8bf*)(sBt1 + bt_n * 32 + bt_kh)     = pk0;
      *(v8bf*)(sBt1 + bt_n * 32 + bt_kh + 8) = pk1;
      if (DUAL) {
        const float* src2 = W2 + (long)(k0 + bt_kh) * ldw + n0 + bt_n;
        v8bf qk0, qk1;
#pragma unroll
        for (int j = 0; j < 8; ++j) qk0[j] = (__bf16)src2[(long)j * ldw];
#pragma unroll
        for (int j = 0; j < 8; ++j) qk1[j] = (__bf16)src2[(long)(8 + j) * ldw];
        *(v8bf*)(sBt2 + bt_n * 32 + bt_kh)     = qk0;
        *(v8bf*)(sBt2 + bt_n * 32 + bt_kh + 8) = qk1;
      }
    }
    __syncthreads();

    v16bf af = frag_from(sA + (wm * 16 + lr) * 32, half);
#pragma unroll
    for (int ns = 0; ns < 4; ++ns) {
      int nc = wn * 64 + ns * 16 + lr;
      v16bf bf1 = frag_from(sBt1 + nc * 32, half);
      acc[ns] = wmma_bf16(af, bf1, acc[ns]);
      if (DUAL) {
        v16bf bf2 = frag_from(sBt2 + nc * 32, half);
        acc2[ns] = wmma_bf16(af, bf2, acc2[ns]);
      }
    }
    __syncthreads();
  }

  // ---- epilogue (C layout: VGPR j -> row j + half*8, lane -> col lr) ----
#pragma unroll
  for (int ns = 0; ns < 4; ++ns) {
#pragma unroll
    for (int j = 0; j < 8; ++j) {
      int m = m0 + wm * 16 + j + half * 8;
      if (m >= Mrows) continue;
      int bb = m / Ltok;
      int tt = tok_off + (m % Ltok);
      long g = (long)bb * kL + tt;
      int n = n0 + wn * 64 + ns * 16 + lr;
      float c = acc[ns][j];
      if (DUAL) {
        float u = acc2[ns][j];
        outB[g * (long)ldo + n] = (__bf16)(gelu_tanh(c) * u);
      } else if (mode == 0) {
        outF[g * (long)ldo + n] = c;
      } else if (mode == 1) {
        outF[g * (long)ldo + n] += c;
      } else {  // mode 3: transposed bf16 store vT[b][n][t]
        outB[((long)bb * kHD + n) * kL + tt] = (__bf16)c;
      }
    }
  }
}

// ---------------------------------------------------------------------------
// RoPE: q rotated in place (f32); k rotated and converted to bf16 (kout),
// which is exactly the attention sK tile layout.
// ---------------------------------------------------------------------------
__global__ void rope_kernel(float* __restrict__ q, const float* __restrict__ kin,
                            __bf16* __restrict__ kout) {
  long idx = (long)blockIdx.x * blockDim.x + threadIdx.x;
  long total = (long)kB * kL * (kH + 1) * (kHD / 2);
  if (idx >= total) return;
  int i = (int)(idx % (kHD / 2));
  long r = idx / (kHD / 2);
  int head = (int)(r % (kH + 1));
  long bt = r / (kH + 1);
  int t = (int)(bt % kL);
  float inv = __expf(-((float)(2 * i) / (float)kHD) * 9.2103403719761836f);
  float ang = (float)t * inv;
  float c = __cosf(ang), s = __sinf(ang);
  if (head < kH) {
    float* p = q + bt * (long)(kH * kHD) + head * kHD;
    float a = p[i];
    float b = p[i + kHD / 2];
    p[i]           = a * c - b * s;
    p[i + kHD / 2] = b * c + a * s;
  } else {
    const float* p = kin + bt * (long)kHD;
    __bf16* o = kout + bt * (long)kHD;
    float a = p[i];
    float b = p[i + kHD / 2];
    o[i]           = (__bf16)(a * c - b * s);
    o[i + kHD / 2] = (__bf16)(b * c + a * s);
  }
}

// ---------------------------------------------------------------------------
// Flash attention. Block = 8 waves = 128 q rows of one (b, h).
// K tile [key][hd] and V tile [hd][key] are bf16 in global memory already, so
// both stage as raw TDM tile loads with hardware OOB zero-fill; all WMMA
// fragments are 2 x ds_load_b128.
// ---------------------------------------------------------------------------
__global__ __launch_bounds__(256) void attn_kernel(const float* __restrict__ qbuf,
                                                   const __bf16* __restrict__ kbb,
                                                   const __bf16* __restrict__ vT,
                                                   __bf16* __restrict__ attb) {
  __shared__ alignas(16) __bf16 sK[32 * 256];   // [key][hd]
  __shared__ alignas(16) __bf16 sV[256 * 32];   // [hd][key]
  __shared__ alignas(16) __bf16 sP[8][16 * 32]; // per-wave P tile

  const int tid  = threadIdx.x;
  const int lane = tid & 31;
  const int wid  = tid >> 5;
  const int half = lane >> 4;
  const int lr   = lane & 15;

  const int nQ = (kL + 127) / 128;   // 7
  int qt_blk = blockIdx.x % nQ;
  int bh = blockIdx.x / nQ;
  int h = bh % kH;
  int b = bh / kH;
  int q0 = qt_blk * 128 + wid * 16;

  // ---- Q fragments (vectorized f32 loads -> bf16, once per wave) ----
  v16bf qf[8];
  {
    int myrow = q0 + lr;
    bool qv = myrow < kL;
    const float* qrow =
        qbuf + ((long)b * kL + (qv ? myrow : 0)) * (long)(kH * kHD) + h * kHD;
    v8bf z8;
#pragma unroll
    for (int j = 0; j < 8; ++j) z8[j] = (__bf16)0.f;
#pragma unroll
    for (int kc = 0; kc < 8; ++kc) {
      v8bf lo = qv ? cvt8(qrow + kc * 32 + half * 8) : z8;
      v8bf hi = qv ? cvt8(qrow + kc * 32 + 16 + half * 8) : z8;
      qf[kc] = __builtin_shufflevector(lo, hi, 0, 1, 2, 3, 4, 5, 6, 7, 8, 9, 10,
                                       11, 12, 13, 14, 15);
    }
  }

  v8f zf = {};
  v8f oacc[16];
#pragma unroll
  for (int i = 0; i < 16; ++i) oacc[i] = zf;
  float mrow[8], lrow[8];
#pragma unroll
  for (int j = 0; j < 8; ++j) { mrow[j] = kNegBig; lrow[j] = 0.f; }

  for (int kt0 = 0; kt0 < kL; kt0 += 32) {
    __syncthreads();   // previous iteration's fragment reads complete
#if ATHENA_HAVE_TDM
    if (wid == 0) {
      // K tile: 32 keys x 256 hd; rows past L zero-filled via tensor_dim1.
      tdm_load_2d(kbb + ((long)b * kL + kt0) * kHD, (unsigned)(size_t)(&sK[0]),
                  (unsigned)kHD, (unsigned)(kL - kt0), (unsigned)kHD, 32u,
                  (unsigned)kHD);
      // V tile: 256 hd rows x 32 keys; keys past L zero-filled via tensor_dim0.
      tdm_load_2d(vT + (long)b * kHD * kL + kt0, (unsigned)(size_t)(&sV[0]),
                  (unsigned)(kL - kt0), (unsigned)kHD, 32u, (unsigned)kHD,
                  (unsigned)kL);
      __builtin_amdgcn_s_wait_tensorcnt(0);
    }
#else
    // manual fallback: bf16 copies with guards
    {
      int key = tid >> 3, seg = (tid & 7) * 32;
      int gk = kt0 + key;
      v8bf z8;
#pragma unroll
      for (int j = 0; j < 8; ++j) z8[j] = (__bf16)0.f;
      const __bf16* src = kbb + ((long)b * kL + (gk < kL ? gk : 0)) * kHD + seg;
#pragma unroll
      for (int r = 0; r < 4; ++r) {
        v8bf pk = (gk < kL) ? *(const v8bf*)(src + r * 8) : z8;
        *(v8bf*)(sK + key * 256 + seg + r * 8) = pk;
      }
      const __bf16* vsrc = vT + ((long)b * kHD + tid) * kL;
#pragma unroll
      for (int r = 0; r < 4; ++r) {
        v8bf pk;
#pragma unroll
        for (int j = 0; j < 8; ++j) {
          int kk = kt0 + r * 8 + j;
          pk[j] = (kk < kL) ? vsrc[kk] : (__bf16)0.f;
        }
        *(v8bf*)(sV + tid * 32 + r * 8) = pk;
      }
    }
#endif
    __syncthreads();

    // ---- S = Q K^T over 32 keys ----
    v8f sc0 = zf, sc1 = zf;
#pragma unroll
    for (int kc = 0; kc < 8; ++kc) {
      v16bf kf = frag_from(sK + lr * 256 + kc * 32, half);
      sc0 = wmma_bf16(qf[kc], kf, sc0);
    }
#pragma unroll
    for (int kc = 0; kc < 8; ++kc) {
      v16bf kf = frag_from(sK + (16 + lr) * 256 + kc * 32, half);
      sc1 = wmma_bf16(qf[kc], kf, sc1);
    }

    // ---- mask + online softmax ----
#pragma unroll
    for (int j = 0; j < 8; ++j) {
      int row = j + half * 8;
      int qt = q0 + row;
      int k0i = kt0 + lr;
      int k1i = kt0 + 16 + lr;
      bool a0 = (k0i < kL) && ((k0i < kL1) || (k0i <= qt));
      bool a1 = (k1i < kL) && ((k1i < kL1) || (k1i <= qt));
      float v0 = a0 ? sc0[j] * kScale : kNegBig;
      float v1 = a1 ? sc1[j] * kScale : kNegBig;
      float mx = fmaxf(v0, v1);
#pragma unroll
      for (int d = 1; d < 16; d <<= 1) mx = fmaxf(mx, __shfl_xor(mx, d, 32));
      float mnew = fmaxf(mrow[j], mx);
      float alpha = __expf(mrow[j] - mnew);
      float p0 = __expf(v0 - mnew);
      float p1 = __expf(v1 - mnew);
      float ts = p0 + p1;
#pragma unroll
      for (int d = 1; d < 16; d <<= 1) ts += __shfl_xor(ts, d, 32);
      lrow[j] = lrow[j] * alpha + ts;
      mrow[j] = mnew;
#pragma unroll
      for (int hc = 0; hc < 16; ++hc) oacc[hc][j] *= alpha;
      sP[wid][row * 32 + lr]      = (__bf16)p0;
      sP[wid][row * 32 + 16 + lr] = (__bf16)p1;
    }
    // per-wave LDS region: per-wave DS ordering, just drain DS counter
    asm volatile("s_wait_dscnt 0" ::: "memory");
    v16bf pf = frag_from(&sP[wid][lr * 32], half);

    // ---- O += P V ----
#pragma unroll
    for (int hc = 0; hc < 16; ++hc) {
      v16bf vf = frag_from(sV + (hc * 16 + lr) * 32, half);
      oacc[hc] = wmma_bf16(pf, vf, oacc[hc]);
    }
  }

  // ---- normalize + store att (B, L, H*HD) bf16 ----
#pragma unroll
  for (int hc = 0; hc < 16; ++hc) {
#pragma unroll
    for (int j = 0; j < 8; ++j) {
      int row = j + half * 8;
      int qt = q0 + row;
      if (qt < kL) {
        float val = oacc[hc][j] / lrow[j];
        attb[((long)b * kL + qt) * (long)(kH * kHD) + h * kHD + hc * 16 + lr] =
            (__bf16)val;
      }
    }
  }
}

// ---------------------------------------------------------------------------
extern "C" void kernel_launch(void* const* d_in, const int* in_sizes, int n_in,
                              void* d_out, int out_size, void* d_ws, size_t ws_size,
                              hipStream_t stream) {
  const float* prefix = (const float*)d_in[0];
  const float* suffix = (const float*)d_in[1];
  const float* p_ln1  = (const float*)d_in[4];
  const float* p_q    = (const float*)d_in[5];
  const float* p_k    = (const float*)d_in[6];
  const float* p_v    = (const float*)d_in[7];
  const float* p_o    = (const float*)d_in[8];
  const float* p_ln2  = (const float*)d_in[9];
  const float* p_gate = (const float*)d_in[10];
  const float* p_up   = (const float*)d_in[11];
  const float* p_down = (const float*)d_in[12];
  const float* e_ln1  = (const float*)d_in[13];
  const float* e_q    = (const float*)d_in[14];
  const float* e_k    = (const float*)d_in[15];
  const float* e_v    = (const float*)d_in[16];
  const float* e_o    = (const float*)d_in[17];
  const float* e_ln2  = (const float*)d_in[18];
  const float* e_gate = (const float*)d_in[19];
  const float* e_up   = (const float*)d_in[20];
  const float* e_down = (const float*)d_in[21];

  char* wp = (char*)d_ws;
  auto alloc = [&](size_t bytes) -> void* {
    void* p = (void*)wp;
    wp += (bytes + 255) & ~(size_t)255;
    return p;
  };
  const size_t tokD = (size_t)kB * kL * kD;
  float*  h    = (float*)alloc(tokD * 4);
  __bf16* nb   = (__bf16*)alloc(tokD * 2);
  float*  qb   = (float*)alloc(tokD * 4);
  float*  kb   = (float*)alloc((size_t)kB * kL * kHD * 4);   // pre-rope K (f32)
  __bf16* kbb  = (__bf16*)alloc((size_t)kB * kL * kHD * 2);  // post-rope K (bf16)
  __bf16* vT   = (__bf16*)alloc((size_t)kB * kHD * kL * 2);  // V transposed (bf16)
  __bf16* ab   = (__bf16*)alloc(tokD * 2);
  __bf16* actb = (__bf16*)alloc((size_t)kB * kL * kMV * 2);

  const int MP = kB * kL1;        // 3072
  const int MS = kB * kL2;        // 200
  const int GP = MP / 64;         // 48
  const int GS = (MS + 63) / 64;  // 4

  {
    long total = (long)tokD;
    concat_init_kernel<<<dim3((unsigned)((total + 255) / 256)), dim3(256), 0,
                         stream>>>(prefix, suffix, h);
  }

  for (int layer = 0; layer < 2; ++layer) {
    const float* pw_ln1  = p_ln1  + (size_t)layer * kD;
    const float* pw_q    = p_q    + (size_t)layer * kD * (kH * kHD);
    const float* pw_k    = p_k    + (size_t)layer * kD * kHD;
    const float* pw_v    = p_v    + (size_t)layer * kD * kHD;
    const float* pw_o    = p_o    + (size_t)layer * (kH * kHD) * kD;
    const float* pw_ln2  = p_ln2  + (size_t)layer * kD;
    const float* pw_gate = p_gate + (size_t)layer * kD * kMV;
    const float* pw_up   = p_up   + (size_t)layer * kD * kMV;
    const float* pw_down = p_down + (size_t)layer * kMV * kD;
    const float* ew_ln1  = e_ln1  + (size_t)layer * kD;
    const float* ew_q    = e_q    + (size_t)layer * kD * (kH * kHD);
    const float* ew_k    = e_k    + (size_t)layer * kD * kHD;
    const float* ew_v    = e_v    + (size_t)layer * kD * kHD;
    const float* ew_o    = e_o    + (size_t)layer * (kH * kHD) * kD;
    const float* ew_ln2  = e_ln2  + (size_t)layer * kD;
    const float* ew_gate = e_gate + (size_t)layer * kD * kME;
    const float* ew_up   = e_up   + (size_t)layer * kD * kME;
    const float* ew_down = e_down + (size_t)layer * kME * kD;

    rmsnorm_kernel<<<dim3(kB * kL), dim3(256), 0, stream>>>(h, pw_ln1, ew_ln1, nb);

    // Q/K projections (f32 out; K gets RoPE'd then bf16-converted)
    gemm_bf16_kernel<false><<<dim3(GP, 16), dim3(256), 0, stream>>>(
        nb, kD, pw_q, nullptr, kH * kHD, qb, nullptr, kH * kHD, MP, kD, kL1, 0, 1, 0);
    gemm_bf16_kernel<false><<<dim3(GP, 2), dim3(256), 0, stream>>>(
        nb, kD, pw_k, nullptr, kHD, kb, nullptr, kHD, MP, kD, kL1, 0, 1, 0);
    gemm_bf16_kernel<false><<<dim3(GS, 16), dim3(256), 0, stream>>>(
        nb, kD, ew_q, nullptr, kH * kHD, qb, nullptr, kH * kHD, MS, kD, kL2, kL1, 0, 0);
    gemm_bf16_kernel<false><<<dim3(GS, 2), dim3(256), 0, stream>>>(
        nb, kD, ew_k, nullptr, kHD, kb, nullptr, kHD, MS, kD, kL2, kL1, 0, 0);
    // V projections: bf16 transposed store straight into attention layout
    gemm_bf16_kernel<false><<<dim3(GP, 2), dim3(256), 0, stream>>>(
        nb, kD, pw_v, nullptr, kHD, nullptr, vT, kL, MP, kD, kL1, 0, 1, 3);
    gemm_bf16_kernel<false><<<dim3(GS, 2), dim3(256), 0, stream>>>(
        nb, kD, ew_v, nullptr, kHD, nullptr, vT, kL, MS, kD, kL2, kL1, 0, 3);

    {
      long total = (long)kB * kL * (kH + 1) * (kHD / 2);
      rope_kernel<<<dim3((unsigned)((total + 255) / 256)), dim3(256), 0, stream>>>(
          qb, kb, kbb);
    }

    {
      int nQ = (kL + 127) / 128;
      attn_kernel<<<dim3(kB * kH * nQ), dim3(256), 0, stream>>>(qb, kbb, vT, ab);
    }

    // O projection + residual
    gemm_bf16_kernel<false><<<dim3(GP, 16), dim3(256), 0, stream>>>(
        ab, kH * kHD, pw_o, nullptr, kD, h, nullptr, kD, MP, kH * kHD, kL1, 0, 1, 1);
    gemm_bf16_kernel<false><<<dim3(GS, 16), dim3(256), 0, stream>>>(
        ab, kH * kHD, ew_o, nullptr, kD, h, nullptr, kD, MS, kH * kHD, kL2, kL1, 0, 1);

    rmsnorm_kernel<<<dim3(kB * kL), dim3(256), 0, stream>>>(h, pw_ln2, ew_ln2, nb);

    // fused gate & up
    gemm_bf16_kernel<true><<<dim3(GP, kMV / 128), dim3(256), 0, stream>>>(
        nb, kD, pw_gate, pw_up, kMV, nullptr, actb, kMV, MP, kD, kL1, 0, 1, 2);
    gemm_bf16_kernel<true><<<dim3(GS, kME / 128), dim3(256), 0, stream>>>(
        nb, kD, ew_gate, ew_up, kME, nullptr, actb, kMV, MS, kD, kL2, kL1, 0, 2);

    // down projection + residual
    gemm_bf16_kernel<false><<<dim3(GP, 16), dim3(256), 0, stream>>>(
        actb, kMV, pw_down, nullptr, kD, h, nullptr, kD, MP, kMV, kL1, 0, 1, 1);
    gemm_bf16_kernel<false><<<dim3(GS, 16), dim3(256), 0, stream>>>(
        actb, kMV, ew_down, nullptr, kD, h, nullptr, kD, MS, kME, kL2, kL1, 0, 1);
  }

  hipMemcpyAsync(d_out, h, tokD * sizeof(float), hipMemcpyDeviceToDevice, stream);
}
